// DeformableExtractor_46591805227604
// MI455X (gfx1250) — matrix operand order repaired and checked by reading, not compiled, not run on hardware
//
#include <hip/hip_runtime.h>
#include <hip/hip_bf16.h>

// ---------------------------------------------------------------------------
// DeformableExtractor on gfx1250 (MI455X), wave32 + WMMA f16->f32.
//   k0: weight cast f32 -> f16  ([32][576] W_om padded, [64][576] W_dcn)
//   k1: conv3x3(3->64)+PReLU -> feat(f32); conv3x3(6->64)+PReLU -> off_feat(f16)
//   k2: im2col GEMM K=576, M=32(27), N=16 pixels/WG via v_wmma_f32_16x16x32_f16
//       -> offset[18] (identity on om[0:18]) and mask[9] = sigmoid(om[18:27])
//   k3: DCNv2 bilinear gather -> LDS columns (16x576 f16), GEMM K=576, M=64,
//       N=16 pixels/WG via WMMA, + bias -> out
// ---------------------------------------------------------------------------

typedef _Float16 v8h  __attribute__((ext_vector_type(8)));
typedef _Float16 v16h __attribute__((ext_vector_type(16)));
typedef float    v8f  __attribute__((ext_vector_type(8)));

#define NTOT 40            // B*T
#define HH   64
#define WW   64
#define HW   4096
#define FCH  64
#define COLK 576           // 64 channels * 9 taps
#define CSTR 584           // padded LDS row stride (halfs); 584*2 % 16 == 0

// A-matrix 16x32 f16 fragment (ISA 7.12.2): lane<16 -> row M=lane, K={k0+0..7, k0+16..23}
//                                           lane>=16 -> row M=lane-16, K={k0+8..15, k0+24..31}
__device__ inline v16h load_fragA(const _Float16* __restrict__ row, int k0, bool hi) {
  const v8h* p = (const v8h*)(row + k0 + (hi ? 8 : 0));
  v8h a = p[0];
  v8h b = p[2];   // +16 halfs
  return __builtin_shufflevector(a, b, 0,1,2,3,4,5,6,7,8,9,10,11,12,13,14,15);
}

// B-matrix 32x16 f16 fragment: lane<16 -> col N=lane, K={k0+0..7, k0+16..23}
//                              lane>=16 -> col N=lane-16, K={k0+8..15, k0+24..31}
__device__ inline v16h load_fragB(const _Float16* __restrict__ col, int k0, bool hi) {
  const v8h* p = (const v8h*)(col + k0 + (hi ? 8 : 0));
  v8h a = p[0];
  v8h b = p[2];
  return __builtin_shufflevector(a, b, 0,1,2,3,4,5,6,7,8,9,10,11,12,13,14,15);
}

// ---------------------------------------------------------------------------
__global__ void k0_prep_weights(const float* __restrict__ Wom,
                                const float* __restrict__ Wdcn,
                                _Float16* __restrict__ wom16,   // [32][576]
                                _Float16* __restrict__ wdcn16)  // [64][576]
{
  int i = blockIdx.x * blockDim.x + threadIdx.x;
  if (i < 32 * COLK) {
    int o = i / COLK;
    wom16[i] = (o < 27) ? (_Float16)Wom[i] : (_Float16)0.f;  // OIHW flatten == [o][c*9+tap]
  }
  if (i < 64 * COLK) {
    wdcn16[i] = (_Float16)Wdcn[i];
  }
}

// ---------------------------------------------------------------------------
// Per-pixel small-channel convs: feat = prelu(conv3x3(xf, Wi)+bi, ai)
//                                off_feat = prelu(conv3x3([ref0,xf], Wo)+bo, ao)
__global__ void k1_features(const float* __restrict__ x,
                            const float* __restrict__ Wi, const float* __restrict__ bi,
                            const float* __restrict__ ai,
                            const float* __restrict__ Wo, const float* __restrict__ bo,
                            const float* __restrict__ ao,
                            float* __restrict__ feat,          // [40][64][4096] f32
                            _Float16* __restrict__ offf)       // [40][64][4096] f16
{
  int pix = blockIdx.x * blockDim.x + threadIdx.x;
  if (pix >= NTOT * HW) return;
  int img = pix >> 12;
  int p   = pix & (HW - 1);
  int y = p >> 6, xx = p & 63;
  int ref = (img / 5) * 5;   // frame 0 of this batch element

  float win[6][9];           // ch 0..2 = ref0, ch 3..5 = frame t
#pragma unroll
  for (int c = 0; c < 3; ++c) {
#pragma unroll
    for (int tap = 0; tap < 9; ++tap) {
      int yy = y + tap / 3 - 1;
      int xq = xx + tap % 3 - 1;
      bool ok = (yy >= 0) & (yy < HH) & (xq >= 0) & (xq < WW);
      int at = ((img * 3 + c) << 12) + (yy << 6) + xq;
      int ar = ((ref * 3 + c) << 12) + (yy << 6) + xq;
      win[3 + c][tap] = ok ? x[at] : 0.f;
      win[c][tap]     = ok ? x[ar] : 0.f;
    }
  }

  for (int o = 0; o < FCH; ++o) {
    float acc1 = bi[o];
    const float* w1 = Wi + o * 27;
#pragma unroll
    for (int c = 0; c < 3; ++c)
#pragma unroll
      for (int tap = 0; tap < 9; ++tap)
        acc1 = fmaf(w1[c * 9 + tap], win[3 + c][tap], acc1);
    float a1 = ai[o];
    acc1 = (acc1 >= 0.f) ? acc1 : a1 * acc1;
    feat[((img * FCH + o) << 12) + p] = acc1;

    float acc2 = bo[o];
    const float* w2 = Wo + o * 54;
#pragma unroll
    for (int c6 = 0; c6 < 6; ++c6)
#pragma unroll
      for (int tap = 0; tap < 9; ++tap)
        acc2 = fmaf(w2[c6 * 9 + tap], win[c6][tap], acc2);
    float a2 = ao[o];
    acc2 = (acc2 >= 0.f) ? acc2 : a2 * acc2;
    offf[((img * FCH + o) << 12) + p] = (_Float16)acc2;
  }
}

// ---------------------------------------------------------------------------
// conv3x3(off_feat, W_om) as implicit GEMM; WG = 16 pixels, 2 waves = 32 rows.
__global__ void k2_offsetmask(const _Float16* __restrict__ offf,
                              const _Float16* __restrict__ Wom16,  // [32][576]
                              const float* __restrict__ bom,       // [27]
                              float* __restrict__ offsetBuf,       // [40][18][4096]
                              float* __restrict__ maskBuf)         // [40][9][4096]
{
  __shared__ __align__(16) _Float16 colA[16 * CSTR];

  int tile = blockIdx.x;
  int img  = tile >> 8;          // 256 tiles / image
  int rem  = tile & 255;
  int y    = rem >> 2;
  int x0   = (rem & 3) << 4;

  int lane = threadIdx.x & 31;
  int wave = threadIdx.x >> 5;   // 0..1
  int row  = lane & 15;
  bool hi  = lane >= 16;
  const _Float16* Arow = Wom16 + (wave * 16 + row) * COLK;
  __builtin_prefetch(Arow, 0, 1);   // global_prefetch_b8

  // im2col: colA[pixel][c*9+tap] = off_feat[c, y+ky-1, x+kx-1] (zero pad)
  for (int idx = threadIdx.x; idx < 16 * COLK; idx += blockDim.x) {
    int pixel = idx / COLK;
    int k     = idx - pixel * COLK;
    int c     = k / 9;
    int tap   = k - c * 9;
    int yy = y + tap / 3 - 1;
    int xq = x0 + pixel + tap % 3 - 1;
    _Float16 v = (_Float16)0.f;
    if (yy >= 0 && yy < HH && xq >= 0 && xq < WW)
      v = offf[((img * FCH + c) << 12) + (yy << 6) + xq];
    colA[pixel * CSTR + k] = v;
  }
  __syncthreads();

  const _Float16* Brow = colA + row * CSTR;
  v8f acc = {0.f, 0.f, 0.f, 0.f, 0.f, 0.f, 0.f, 0.f};
#pragma unroll 2
  for (int kb = 0; kb < COLK / 32; ++kb) {
    v16h a = load_fragA(Arow, kb * 32, hi);
    v16h b = load_fragB(Brow, kb * 32, hi);
    acc = __builtin_amdgcn_wmma_f32_16x16x32_f16(false, a, false, b,
                                                 (short)0, acc, false, false);
  }

  // D layout: VGPR r -> M=r (lanes 0-15) / M=r+8 (lanes 16-31); N = lane&15
  int xn = x0 + row;
#pragma unroll
  for (int r = 0; r < 8; ++r) {
    int m = wave * 16 + r + (hi ? 8 : 0);
    if (m < 27) {
      float v = acc[r] + bom[m];
      if (m < 18)
        offsetBuf[((img * 18 + m) << 12) + (y << 6) + xn] = v;
      else
        maskBuf[((img * 9 + (m - 18)) << 12) + (y << 6) + xn] =
            1.f / (1.f + __expf(-v));
    }
  }
}

// ---------------------------------------------------------------------------
// DCNv2: bilinear-gather columns into LDS, then 64x16x576 GEMM via WMMA.
// WG = 16 pixels, 4 waves = 64 output channels.
__global__ void k3_dcn(const float* __restrict__ feat,
                       const float* __restrict__ offsetBuf,
                       const float* __restrict__ maskBuf,
                       const _Float16* __restrict__ Wdcn16,  // [64][576]
                       const float* __restrict__ bdcn,
                       float* __restrict__ out)              // [40][64][4096]
{
  __shared__ __align__(16) _Float16 colV[16 * CSTR];

  int tile = blockIdx.x;
  int img  = tile >> 8;
  int rem  = tile & 255;
  int y    = rem >> 2;
  int x0   = (rem & 3) << 4;

  int lane = threadIdx.x & 31;
  int wave = threadIdx.x >> 5;   // 0..3
  int row  = lane & 15;
  bool hi  = lane >= 16;
  const _Float16* Arow = Wdcn16 + (wave * 16 + row) * COLK;
  __builtin_prefetch(Arow, 0, 1);

  const float* fimg = feat + (size_t)(img * FCH) * HW;

  // 16 pixels * 9 taps = 144 (pixel,tap) sampling jobs
  for (int pr = threadIdx.x; pr < 144; pr += blockDim.x) {
    int pixel = pr / 9;
    int tap   = pr - pixel * 9;
    int xq    = x0 + pixel;
    int base  = (y << 6) + xq;
    float dy = offsetBuf[((img * 18 + 2 * tap)     << 12) + base];
    float dx = offsetBuf[((img * 18 + 2 * tap + 1) << 12) + base];
    float mv = maskBuf[((img * 9 + tap) << 12) + base];

    float py = (float)y  + (float)(tap / 3 - 1) + dy;
    float px = (float)xq + (float)(tap % 3 - 1) + dx;
    float y0f = floorf(py), x0f = floorf(px);
    float wy = py - y0f, wx = px - x0f;
    int yi = (int)y0f, xi = (int)x0f;

    bool oky0 = (yi     >= 0) & (yi     < HH);
    bool oky1 = (yi + 1 >= 0) & (yi + 1 < HH);
    bool okx0 = (xi     >= 0) & (xi     < WW);
    bool okx1 = (xi + 1 >= 0) & (xi + 1 < WW);
    // fold validity into corner weights (matches reference clip * valid-mask)
    float w00 = (oky0 & okx0) ? (1.f - wy) * (1.f - wx) : 0.f;
    float w01 = (oky0 & okx1) ? (1.f - wy) * wx         : 0.f;
    float w10 = (oky1 & okx0) ? wy * (1.f - wx)         : 0.f;
    float w11 = (oky1 & okx1) ? wy * wx                 : 0.f;
    int yc0 = min(max(yi, 0), HH - 1),     yc1 = min(max(yi + 1, 0), HH - 1);
    int xc0 = min(max(xi, 0), WW - 1),     xc1 = min(max(xi + 1, 0), WW - 1);
    int i00 = (yc0 << 6) + xc0, i01 = (yc0 << 6) + xc1;
    int i10 = (yc1 << 6) + xc0, i11 = (yc1 << 6) + xc1;

    for (int c = 0; c < FCH; ++c) {
      const float* fc = fimg + (c << 12);
      float s = w00 * fc[i00] + w01 * fc[i01] + w10 * fc[i10] + w11 * fc[i11];
      colV[pixel * CSTR + c * 9 + tap] = (_Float16)(s * mv);
    }
  }
  __syncthreads();

  const _Float16* Brow = colV + row * CSTR;
  v8f acc = {0.f, 0.f, 0.f, 0.f, 0.f, 0.f, 0.f, 0.f};
#pragma unroll 2
  for (int kb = 0; kb < COLK / 32; ++kb) {
    v16h a = load_fragA(Arow, kb * 32, hi);
    v16h b = load_fragB(Brow, kb * 32, hi);
    acc = __builtin_amdgcn_wmma_f32_16x16x32_f16(false, a, false, b,
                                                 (short)0, acc, false, false);
  }

  int xn = x0 + row;
#pragma unroll
  for (int r = 0; r < 8; ++r) {
    int m = wave * 16 + r + (hi ? 8 : 0);
    out[((size_t)(img * FCH + m) << 12) + (y << 6) + xn] = acc[r] + bdcn[m];
  }
}

// ---------------------------------------------------------------------------
extern "C" void kernel_launch(void* const* d_in, const int* in_sizes, int n_in,
                              void* d_out, int out_size, void* d_ws, size_t ws_size,
                              hipStream_t stream) {
  (void)in_sizes; (void)n_in; (void)out_size; (void)ws_size;

  const float* x      = (const float*)d_in[0];
  // d_in[1] = flow (unused)
  const float* W_init = (const float*)d_in[2];
  const float* b_init = (const float*)d_in[3];
  const float* a_init = (const float*)d_in[4];
  const float* W_off  = (const float*)d_in[5];
  const float* b_off  = (const float*)d_in[6];
  const float* a_off  = (const float*)d_in[7];
  const float* W_om   = (const float*)d_in[8];
  const float* b_om   = (const float*)d_in[9];
  const float* W_dcn  = (const float*)d_in[10];
  const float* b_dcn  = (const float*)d_in[11];
  float* out = (float*)d_out;

  // workspace carve-up (all 256B aligned)
  char* ws = (char*)d_ws;
  size_t off = 0;
  float*    feat      = (float*)(ws + off);    off += (size_t)NTOT * FCH * HW * 4;  // 41.9 MB
  _Float16* offf      = (_Float16*)(ws + off); off += (size_t)NTOT * FCH * HW * 2;  // 21.0 MB
  float*    offsetBuf = (float*)(ws + off);    off += (size_t)NTOT * 18  * HW * 4;  // 11.8 MB
  float*    maskBuf   = (float*)(ws + off);    off += (size_t)NTOT * 9   * HW * 4;  //  5.9 MB
  _Float16* wom16     = (_Float16*)(ws + off); off += (size_t)32 * COLK * 2;
  _Float16* wdcn16    = (_Float16*)(ws + off); off += (size_t)64 * COLK * 2;

  // k0: weight casts
  k0_prep_weights<<<(64 * COLK + 255) / 256, 256, 0, stream>>>(W_om, W_dcn, wom16, wdcn16);

  // k1: per-pixel features (3->64 and 6->64 convs + PReLU)
  k1_features<<<(NTOT * HW + 255) / 256, 256, 0, stream>>>(
      x, W_init, b_init, a_init, W_off, b_off, a_off, feat, offf);

  // k2: offset/mask head GEMM (WMMA), 256 tiles per image
  k2_offsetmask<<<NTOT * 256, 64, 0, stream>>>(offf, wom16, b_om, offsetBuf, maskBuf);

  // k3: DCNv2 gather + GEMM (WMMA)
  k3_dcn<<<NTOT * 256, 128, 0, stream>>>(feat, offsetBuf, maskBuf, wdcn16, b_dcn, out);
}
